// InformationAggregationLayer_35948876268448
// MI455X (gfx1250) — matrix operand order repaired and checked by reading, not compiled, not run on hardware
//
#include <hip/hip_runtime.h>
#include <hip/hip_bf16.h>

// ---------------------------------------------------------------------------
// InformationAggregationLayer fused kernel for gfx1250 (MI455X).
// B=8, H=50, S=512, E=128, D=256.
// Round 2: double-buffered GLOBAL_LOAD_ASYNC_TO_LDS_B128 for the hist tiles
// (ASYNCcnt-tracked, overlaps WMMA compute with the next tile's HBM fetch).
// ---------------------------------------------------------------------------

#define IA_B 8
#define IA_H 50
#define IA_S 512
#define IA_E 128
#define IA_D 256

typedef __attribute__((ext_vector_type(16))) __bf16 v16bf;
typedef __attribute__((ext_vector_type(8)))  float  v8f;

__device__ __forceinline__ __bf16 f2bf(float f) {
    union { float f; unsigned u; } x; x.f = f;
    unsigned r = x.u + 0x7fffu + ((x.u >> 16) & 1u);   // round-to-nearest-even
    unsigned short h = (unsigned short)(r >> 16);
    __bf16 b; __builtin_memcpy(&b, &h, 2);
    return b;
}

// A-matrix (16x32 bf16) fragment index decode: packed index p in [0, 16*K)
// laid out as kt tiles of 512 entries, each entry = aFrag[kt*512 + lane*16 + e].
// Per CDNA5 ISA "16-bit A-Matrix 16x32" lane/K table.
__device__ __forceinline__ void decodeA(int p, int& m, int& k) {
    int kt  = p >> 9;
    int rem = p & 511;
    int L   = rem >> 4;      // lane
    int e   = rem & 15;      // element within v16bf
    int j   = e >> 1;        // VGPR pair index
    int hb  = e & 1;         // low/high half of dword
    int g   = L >> 4;        // lane group
    m = L & 15;
    int kin = 2 * j + 8 * g + ((j >= 4) ? 8 : 0) + hb;
    k = kt * 32 + kin;
}

// Async copy of one 16 KB tile (16x256 f32, contiguous) into LDS.
// Each of the 512 threads moves 2x16 bytes; ASYNCcnt += 2 per thread.
// LDS address = VGPR offset plus instruction offset; generic->LDS offset = low 32 bits.
__device__ __forceinline__ void async_tile_load(const float* __restrict__ gsrc,
                                                float* ldst, int tid) {
    unsigned    l0 = (unsigned)(uintptr_t)ldst + (unsigned)(tid * 16);
    const char* g0 = (const char*)gsrc + tid * 16;
    asm volatile("global_load_async_to_lds_b128 %0, %1, off"
                 :: "v"(l0), "v"(g0) : "memory");
    asm volatile("global_load_async_to_lds_b128 %0, %1, off offset:8192"
                 :: "v"(l0), "v"(g0) : "memory");
}

// ---------------------------------------------------------------------------
// Kernel 1: pack wc_w / wh_w / wf_w (row-major f32 [N,K]) into bf16 B-matrix
// (K x N) fragments. B layout (per SWMMAC B-matrix note): lane L holds column
// n = L%16; VGPR j holds K = 2j + 8*(L/16) + (j>=4 ? 8 : 0) + {0,1}.
// Packed tile = 512 bf16; element addr = tile*512 + lane*16 + e.
// wc/wh: 256x256 -> 16 N-tiles x 8 K-tiles, tile id = nt*8 + kt.
// wf:    128x512 ->  8 N-tiles x 16 K-tiles, tile id = nt*16 + kt.
// ---------------------------------------------------------------------------
__global__ void ia_pack_weights(const float* __restrict__ wc_w,
                                const float* __restrict__ wh_w,
                                const float* __restrict__ wf_w,
                                __bf16* __restrict__ wcP,
                                __bf16* __restrict__ whP,
                                __bf16* __restrict__ wfP) {
    int t = blockIdx.x * blockDim.x + threadIdx.x;   // 0 .. 196607
    if (t >= 3 * 65536) return;
    int which = t >> 16;
    int q     = t & 65535;
    int tile  = q >> 9;
    int rem   = q & 511;
    int L  = rem >> 4;
    int e  = rem & 15;
    int j  = e >> 1;
    int hb = e & 1;
    int g  = L >> 4;
    int nl = L & 15;
    int kin = 2 * j + 8 * g + ((j >= 4) ? 8 : 0) + hb;
    if (which == 0) {
        int nt = tile >> 3, kt = tile & 7;
        wcP[q] = f2bf(wc_w[(nt * 16 + nl) * IA_D + kt * 32 + kin]);
    } else if (which == 1) {
        int nt = tile >> 3, kt = tile & 7;
        whP[q] = f2bf(wh_w[(nt * 16 + nl) * IA_D + kt * 32 + kin]);
    } else {
        int nt = tile >> 4, kt = tile & 15;
        wfP[q] = f2bf(wf_w[(nt * 16 + nl) * (2 * IA_D) + kt * 32 + kin]);
    }
}

// ---------------------------------------------------------------------------
// Kernel 2: fused aggregation. One block per (b, s-tile of 16 rows).
// 512 threads = 16 wave32; wave w owns output N-tile w.
// ---------------------------------------------------------------------------
__global__ __launch_bounds__(512)
void ia_main(const float* __restrict__ hist,   // [B,H,S,D]
             const float* __restrict__ cur,    // [B,S,D]
             const float* __restrict__ wc_b,   // [D]
             const float* __restrict__ wh_b,   // [D]
             const float* __restrict__ qt_w,   // [1,D]
             const float* __restrict__ qt_b,   // [1]
             const float* __restrict__ wf_b,   // [E]
             const __bf16* __restrict__ wcP,
             const __bf16* __restrict__ whP,
             const __bf16* __restrict__ wfP,
             float* __restrict__ out)          // [B,S,E]
{
    extern __shared__ char smem[];
    float*  curF    = (float*) (smem);              // 16x256 f32  (16 KB)
    float*  wcOut   = (float*) (smem + 16384);      // 16x256 f32  (16 KB)
    float*  histF0  = (float*) (smem + 32768);      // 16x256 f32  (16 KB)
    float*  histF1  = (float*) (smem + 49152);      // 16x256 f32  (16 KB)
    float*  hsum    = (float*) (smem + 65536);      // 16x256 f32  (16 KB)
    __bf16* aFrag   = (__bf16*)(smem + 81920);      // 16x256 bf16 ( 8 KB)
    float*  alpha   = (float*) (smem + 90112);      // 16 f32
    __bf16* concatA = (__bf16*)(smem + 32768);      // reuses histF0 (16 KB)
    float*  histF[2] = { histF0, histF1 };

    const int tid  = threadIdx.x;
    const int lane = tid & 31;
    const int wave = tid >> 5;
    const int g    = lane >> 4;       // lane group (C/D: rows r vs r+8)
    const int nl   = lane & 15;       // column within N-tile

    const int b  = blockIdx.x >> 5;   // S/16 = 32 tiles per batch
    const int s0 = (blockIdx.x & 31) * 16;

    const float* histBase = hist + (((size_t)b * IA_H) * IA_S + s0) * IA_D;

    // ---- kick off async fetch of h=0 tile immediately ----------------------
    async_tile_load(histBase, histF[0], tid);

    // ---- load current-embedding tile (contiguous 16 KB) --------------------
    {
        const float4* src4 = (const float4*)(cur + ((size_t)b * IA_S + s0) * IA_D);
        float4* dst4 = (float4*)curF;
        dst4[tid]       = src4[tid];
        dst4[tid + 512] = src4[tid + 512];
    }
    __syncthreads();

    // ---- convert cur tile -> bf16 A fragments ------------------------------
    for (int p = tid; p < 4096; p += 512) {
        int m, k; decodeA(p, m, k);
        aFrag[p] = f2bf(curF[m * IA_D + k]);
    }
    __syncthreads();

    // ---- wc_out = cur @ wc_w^T + wc_b  (16 waves x 16-col N-tiles) ---------
    {
        v8f acc = {};
#pragma unroll
        for (int kt = 0; kt < 8; ++kt) {
            v16bf a  = *(const v16bf*)(aFrag + kt * 512 + lane * 16);
            v16bf bm = *(const v16bf*)(wcP + ((size_t)(wave * 8 + kt)) * 512 + lane * 16);
            acc = __builtin_amdgcn_wmma_f32_16x16x32_bf16(false, a, false, bm,
                                                          (short)0, acc, false, false);
        }
        int n = wave * 16 + nl;
        float bias = wc_b[n];
#pragma unroll
        for (int r = 0; r < 8; ++r)
            wcOut[(r + 8 * g) * IA_D + n] = acc[r] + bias;
    }

    // ---- zero history accumulator ------------------------------------------
    for (int p = tid; p < 4096; p += 512) hsum[p] = 0.0f;

    // per-wave invariants for the h-loop
    const int   n_glob = wave * 16 + nl;
    const float qw  = qt_w[n_glob];
    const float whb = wh_b[n_glob];
    const float qtb = qt_b[0];

    // ---- history loop (double-buffered async tiles) ------------------------
    for (int h = 0; h < IA_H; ++h) {
        const int cur_buf = h & 1;

        // prefetch depth 2 into L2
        if (h + 2 < IA_H && tid < 128) {
            const float* far = histBase + (size_t)(h + 2) * IA_S * IA_D;
            __builtin_prefetch(far + tid * 32, 0, 0);
        }

        // issue async fetch of next tile into the other buffer, then retire
        // the current tile (ASYNCcnt completes in order: wait <= 2 keeps the
        // just-issued pair in flight).
        if (h + 1 < IA_H) {
            async_tile_load(histBase + (size_t)(h + 1) * IA_S * IA_D,
                            histF[cur_buf ^ 1], tid);
            asm volatile("s_wait_asynccnt 0x2" ::: "memory");
        } else {
            asm volatile("s_wait_asynccnt 0x0" ::: "memory");
        }

        if (tid < 16) alpha[tid] = qtb;
        __syncthreads();   // tile in LDS + alpha init visible

        const float* hf = histF[cur_buf];

        // convert to bf16 A fragments
        for (int p = tid; p < 4096; p += 512) {
            int m, k; decodeA(p, m, k);
            aFrag[p] = f2bf(hf[m * IA_D + k]);
        }
        __syncthreads();

        // wh_out tile via WMMA (f32 accumulate)
        v8f acc = {};
#pragma unroll
        for (int kt = 0; kt < 8; ++kt) {
            v16bf a  = *(const v16bf*)(aFrag + kt * 512 + lane * 16);
            v16bf bm = *(const v16bf*)(whP + ((size_t)(wave * 8 + kt)) * 512 + lane * 16);
            acc = __builtin_amdgcn_wmma_f32_16x16x32_bf16(false, a, false, bm,
                                                          (short)0, acc, false, false);
        }

        // combined = sigmoid(wc_out + wh_out + wh_b); alpha += qt_w . combined
#pragma unroll
        for (int r = 0; r < 8; ++r) {
            int m = r + 8 * g;
            float v = acc[r] + wcOut[m * IA_D + n_glob] + whb;
            v = 1.0f / (1.0f + __expf(-v));
            float part = v * qw;
            part += __shfl_xor(part, 8, 16);
            part += __shfl_xor(part, 4, 16);
            part += __shfl_xor(part, 2, 16);
            part += __shfl_xor(part, 1, 16);
            if (nl == 0) atomicAdd(&alpha[m], part);   // ds_add_f32
        }
        __syncthreads();

        // hsum += alpha[m] * hist[m][n]
        for (int p = tid; p < 4096; p += 512) {
            int m = p >> 8;
            hsum[p] += alpha[m] * hf[p];
        }
        __syncthreads();
    }

    // ---- build concat [cur | hsum] as bf16 A fragments (16x512, 16 K-tiles) -
    for (int p = tid; p < 8192; p += 512) {
        int m, k; decodeA(p, m, k);
        float v = (k < IA_D) ? curF[m * IA_D + k] : hsum[m * IA_D + (k - IA_D)];
        concatA[p] = f2bf(v);
    }
    __syncthreads();

    // ---- out = concat @ wf_w^T + wf_b  (8 waves, K = 512) -------------------
    if (wave < 8) {
        v8f acc = {};
#pragma unroll
        for (int kt = 0; kt < 16; ++kt) {
            v16bf a  = *(const v16bf*)(concatA + kt * 512 + lane * 16);
            v16bf bm = *(const v16bf*)(wfP + ((size_t)(wave * 16 + kt)) * 512 + lane * 16);
            acc = __builtin_amdgcn_wmma_f32_16x16x32_bf16(false, a, false, bm,
                                                          (short)0, acc, false, false);
        }
        int n = wave * 16 + nl;
        float bias = wf_b[n];
#pragma unroll
        for (int r = 0; r < 8; ++r) {
            int m = r + 8 * g;
            out[((size_t)b * IA_S + s0 + m) * IA_E + n] = acc[r] + bias;
        }
    }
}

// ---------------------------------------------------------------------------
extern "C" void kernel_launch(void* const* d_in, const int* in_sizes, int n_in,
                              void* d_out, int out_size, void* d_ws, size_t ws_size,
                              hipStream_t stream) {
    const float* hist = (const float*)d_in[0];   // [B,H,S,D]
    const float* cur  = (const float*)d_in[1];   // [B,S,D]
    const float* wc_w = (const float*)d_in[2];
    const float* wc_b = (const float*)d_in[3];
    const float* wh_w = (const float*)d_in[4];
    const float* wh_b = (const float*)d_in[5];
    const float* qt_w = (const float*)d_in[6];
    const float* qt_b = (const float*)d_in[7];
    const float* wf_w = (const float*)d_in[8];
    const float* wf_b = (const float*)d_in[9];
    float* out = (float*)d_out;

    __bf16* wcP = (__bf16*)d_ws;          // 65536 bf16
    __bf16* whP = wcP + 65536;            // 65536 bf16
    __bf16* wfP = whP + 65536;            // 65536 bf16

    ia_pack_weights<<<768, 256, 0, stream>>>(wc_w, wh_w, wf_w, wcP, whP, wfP);

    const int blocks = IA_B * (IA_S / 16);            // 256
    const size_t lds = 90112 + 64;                    // ~88 KB dynamic LDS
    ia_main<<<blocks, 512, lds, stream>>>(hist, cur, wc_b, wh_b, qt_w, qt_b,
                                          wf_b, wcP, whP, wfP, out);
}
